// GCNIIppi_75866302316593
// MI455X (gfx1250) — matrix operand order repaired and checked by reading, not compiled, not run on hardware
//
#include <hip/hip_runtime.h>
#include <hip/hip_bf16.h>
#include <math.h>

#define N_NODES 50000
#define N_EDGES 800000
#define NFEAT   512
#define NH      128
#define NL      8
#define NAUX    57
#define NSITE   32
#define ALPHA   0.1f
#define LAMDA   0.5f

typedef __attribute__((ext_vector_type(16))) __bf16 v16bf;
typedef __attribute__((ext_vector_type(8)))  float  v8f;

// ---------------------------------------------------------------------------
// fp32 -> bf16 conversions (RNE) via integer ops: guaranteed lowering.
// ---------------------------------------------------------------------------
__device__ __forceinline__ unsigned short f2bfu(float f) {
    unsigned u = __builtin_bit_cast(unsigned, f);
    return (unsigned short)((u + 0x7FFFu + ((u >> 16) & 1u)) >> 16);
}

// pack two fp32 into one dword of two bf16 (f0 -> low, f1 -> high)
__device__ __forceinline__ unsigned pack2bf(float f0, float f1) {
    unsigned u0 = __builtin_bit_cast(unsigned, f0);
    unsigned u1 = __builtin_bit_cast(unsigned, f1);
    unsigned r0 = (u0 + 0x7FFFu + ((u0 >> 16) & 1u)) >> 16;
    unsigned r1 = (u1 + 0x7FFFu + ((u1 >> 16) & 1u)) & 0xFFFF0000u;
    return r0 | r1;
}

// ===========================================================================
// Weight pre-pack: W[K x 128] fp32 -> bf16 tiles swizzled into the exact
// CDNA5 WMMA B-operand layout, so the GEMM hot loop loads each lane's
// 16-element operand as one contiguous 32-byte vector.
//   packed[((kt*8 + nt)*32 + lane)*16 + j]  =  W[(kt*32 + (j>>3)*16 + (lane>>4)*8 + (j&7))*128
//                                                + nt*16 + (lane&15)]
// One thread per packed element; grid.y selects the matrix slice (for conv_w).
// ===========================================================================
__global__ void pack_weights_kernel(const float* __restrict__ W,
                                    unsigned short* __restrict__ pk,
                                    int elems_per_slice) {
    int t = blockIdx.x * blockDim.x + threadIdx.x;
    size_t slice = (size_t)blockIdx.y * elems_per_slice;
    int j    = t & 15;
    int lane = (t >> 4) & 31;
    int nt   = (t >> 9) & 7;
    int kt   = t >> 12;
    int n  = nt * 16 + (lane & 15);
    int kk = kt * 32 + (j >> 3) * 16 + (lane >> 4) * 8 + (j & 7);
    pk[slice + t] = f2bfu(W[slice + (size_t)kk * NH + n]);
}

// ===========================================================================
// h0 = relu(X[N,512] @ W0[512,128] + b0)
// block = 256 (8 waves), grid = N/16. Wave w handles n-tile w. The A tile
// (shared by all 8 waves) is staged through LDS in bf16 operand layout.
// ===========================================================================
__global__ void input_gemm_kernel(const float* __restrict__ X,
                                  const v16bf* __restrict__ W0pk,
                                  const float* __restrict__ b0,
                                  float* __restrict__ out) {
    __shared__ unsigned aS[256];               // 16x32 bf16 A tile, swizzled
    const v16bf* av = (const v16bf*)aS;

    int lane = threadIdx.x & 31;
    int wid  = threadIdx.x >> 5;               // n-tile 0..7
    int m0   = blockIdx.x * 16;

    // staging role of this thread: elements i0, i0+1 of the A operand
    int i0 = threadIdx.x * 2;                  // 0..510
    int sL = i0 >> 4;                          // operand lane 0..31
    int sj = i0 & 15;                          // element pair start (even)
    int srow = m0 + (sL & 15);
    int skb  = (sL >> 4) * 8;
    int skk  = (sj < 8) ? (skb + sj) : (skb + 8 + sj);
    const float* xrow = X + (size_t)srow * NFEAT + skk;

    v8f c = {};
    for (int kt = 0; kt < NFEAT / 32; ++kt) {
        __syncthreads();                       // previous tile consumed
        const float* xp = xrow + kt * 32;
        float2 f = *(const float2*)xp;
        aS[threadIdx.x] = pack2bf(f.x, f.y);
        if (kt + 1 < NFEAT / 32)
            __builtin_prefetch(xp + 32, 0, 1); // next K tile of this row
        __syncthreads();                       // tile staged

        v16bf a = av[lane];
        v16bf b = W0pk[(kt * 8 + wid) * 32 + lane];
        c = __builtin_amdgcn_wmma_f32_16x16x32_bf16(false, a, false, b,
                                                    (short)0, c, false, false);
    }

    // C/D layout: lanes 0-15 -> N=lane, M=v; lanes 16-31 -> M=8+v
    int ocol = wid * 16 + (lane & 15);
    int orow = m0 + (lane >> 4) * 8;
    float bias = b0[ocol];
#pragma unroll
    for (int v = 0; v < 8; ++v) {
        float r = c[v] + bias;
        out[(size_t)(orow + v) * NH + ocol] = fmaxf(r, 0.0f);
    }
}

// ===========================================================================
// zero fill
// ===========================================================================
__global__ void zero_kernel(float* __restrict__ p) {
    p[(size_t)blockIdx.x * blockDim.x + threadIdx.x] = 0.0f;
}

// ===========================================================================
// SpMM scatter: hi[row] += val * h[col].  One edge per wave; 4 feats/lane.
// hi (25.6 MB) is L2-resident, so fp32 atomics resolve at L2.
// ===========================================================================
__global__ void spmm_kernel(const int* __restrict__ idx,
                            const float* __restrict__ val,
                            const float* __restrict__ h,
                            float* __restrict__ hi) {
    long gid = (long)blockIdx.x * blockDim.x + threadIdx.x;
    int e  = (int)(gid >> 5);
    int ln = (int)(gid & 31);
    int row = idx[e];
    int col = idx[N_EDGES + e];
    float v = val[e];
    const float4 hv = *(const float4*)(h + (size_t)col * NH + ln * 4);
    float* dst = hi + (size_t)row * NH + ln * 4;
    atomicAdd(dst + 0, v * hv.x);
    atomicAdd(dst + 1, v * hv.y);
    atomicAdd(dst + 2, v * hv.z);
    atomicAdd(dst + 3, v * hv.w);
}

// ===========================================================================
// Layer update:
//   support = (1-ALPHA)*hi + ALPHA*h0
//   out = relu(theta*(support @ W) + (1-theta)*support + h_in)
// block = 256 (8 waves = 8 n-tiles), grid = N/16. Support tile staged in LDS.
// ===========================================================================
__global__ void layer_update_kernel(const float* __restrict__ hi,
                                    const float* __restrict__ h0,
                                    const float* __restrict__ hin,
                                    const v16bf* __restrict__ Wpk,
                                    float theta,
                                    float* __restrict__ out) {
    __shared__ unsigned aS[256];
    const v16bf* av = (const v16bf*)aS;

    int lane = threadIdx.x & 31;
    int wid  = threadIdx.x >> 5;
    int m0   = blockIdx.x * 16;

    int i0 = threadIdx.x * 2;
    int sL = i0 >> 4;
    int sj = i0 & 15;
    int srow = m0 + (sL & 15);
    int skb  = (sL >> 4) * 8;
    int skk  = (sj < 8) ? (skb + sj) : (skb + 8 + sj);
    size_t soff = (size_t)srow * NH + skk;

    v8f c = {};
    for (int kt = 0; kt < NH / 32; ++kt) {
        __syncthreads();
        float2 hv = *(const float2*)(hi + soff + kt * 32);
        float2 h0v = *(const float2*)(h0 + soff + kt * 32);
        float s0 = (1.0f - ALPHA) * hv.x + ALPHA * h0v.x;
        float s1 = (1.0f - ALPHA) * hv.y + ALPHA * h0v.y;
        aS[threadIdx.x] = pack2bf(s0, s1);
        __syncthreads();

        v16bf a = av[lane];
        v16bf b = Wpk[(kt * 8 + wid) * 32 + lane];
        c = __builtin_amdgcn_wmma_f32_16x16x32_bf16(false, a, false, b,
                                                    (short)0, c, false, false);
    }

    int ocol = wid * 16 + (lane & 15);
    int orow = m0 + (lane >> 4) * 8;
#pragma unroll
    for (int v = 0; v < 8; ++v) {
        size_t off = (size_t)(orow + v) * NH + ocol;
        float sup = (1.0f - ALPHA) * hi[off] + ALPHA * h0[off];
        float r = theta * c[v] + (1.0f - theta) * sup + hin[off];
        out[off] = fmaxf(r, 0.0f);
    }
}

// ===========================================================================
// li[f] = mean over 32 mutation-site rows of h[:,f].  1 block, 128 threads.
// ===========================================================================
__global__ void gather_mean_kernel(const float* __restrict__ h,
                                   const int* __restrict__ site,
                                   float* __restrict__ dst) {
    int f = threadIdx.x;
    float s = 0.0f;
#pragma unroll
    for (int i = 0; i < NSITE; ++i)
        s += h[(size_t)site[i] * NH + f];
    dst[f] = s * (1.0f / NSITE);
}

// ===========================================================================
// Head MLP. Reads li/wli already in d_out[1..256]; writes differ and the
// final scalar.  1 block, 256 threads.
// ===========================================================================
__global__ void head_kernel(const float* __restrict__ aux,
                            const float* __restrict__ fc_w,
                            const float* __restrict__ fc_b,
                            const float* __restrict__ fc2_w,
                            const float* __restrict__ fc2_b,
                            const float* __restrict__ fc3_w,
                            const float* __restrict__ fc3_b,
                            float* __restrict__ out) {
    __shared__ float d_s[NH + NAUX];
    __shared__ float o1[64];
    __shared__ float o2[32];
    int tid = threadIdx.x;

    if (tid < NH) {
        float li  = out[1 + tid];
        float wli = out[1 + NH + tid];
        float differ = li - wli;
        out[1 + 2 * NH + tid] = differ;          // gbdt[256..383]
        d_s[tid] = fmaxf(differ, 0.0f);
    } else if (tid < NH + NAUX) {
        d_s[tid] = aux[tid - NH];
    }
    __syncthreads();

    if (tid < 64) {
        float s = fc_b[tid];
        for (int k = 0; k < NH + NAUX; ++k) s += d_s[k] * fc_w[k * 64 + tid];
        o1[tid] = fmaxf(s, 0.0f);
    }
    __syncthreads();

    if (tid < 32) {
        float s = fc2_b[tid];
        for (int k = 0; k < 64; ++k) s += o1[k] * fc2_w[k * 32 + tid];
        o2[tid] = fmaxf(s, 0.0f);
    }
    __syncthreads();

    if (tid == 0) {
        float s = fc3_b[0];
        for (int k = 0; k < 32; ++k) s += o2[k] * fc3_w[k];
        out[0] = s;
    }
}

// ===========================================================================
extern "C" void kernel_launch(void* const* d_in, const int* in_sizes, int n_in,
                              void* d_out, int out_size, void* d_ws, size_t ws_size,
                              hipStream_t stream) {
    (void)in_sizes; (void)n_in; (void)out_size; (void)ws_size;

    const float* x            = (const float*)d_in[0];
    const float* wild_feature = (const float*)d_in[1];
    const float* aux          = (const float*)d_in[2];
    const float* adj_val      = (const float*)d_in[3];
    const float* wild_adj_val = (const float*)d_in[4];
    const int*   adj_idx      = (const int*)d_in[5];
    const int*   wild_adj_idx = (const int*)d_in[6];
    const int*   site         = (const int*)d_in[7];
    /* d_in[8] = nodes (unused) */
    const float* w0    = (const float*)d_in[9];
    const float* b0    = (const float*)d_in[10];
    const float* convw = (const float*)d_in[11];
    const float* fc_w  = (const float*)d_in[12];
    const float* fc_b  = (const float*)d_in[13];
    const float* fc2_w = (const float*)d_in[14];
    const float* fc2_b = (const float*)d_in[15];
    const float* fc3_w = (const float*)d_in[16];
    const float* fc3_b = (const float*)d_in[17];
    float* out = (float*)d_out;

    // workspace: 5 x [N, NH] fp32 (h0, wh0, ping, pong, hi) + packed weights
    char* ws = (char*)d_ws;
    size_t mat = (size_t)N_NODES * NH * sizeof(float);   // 25.6 MB
    float* h0   = (float*)(ws + 0 * mat);
    float* wh0  = (float*)(ws + 1 * mat);
    float* ping = (float*)(ws + 2 * mat);
    float* pong = (float*)(ws + 3 * mat);
    float* hi   = (float*)(ws + 4 * mat);
    unsigned short* w0pk   = (unsigned short*)(ws + 5 * mat);
    unsigned short* convpk = w0pk + (size_t)NFEAT * NH;

    // --- pre-pack weights into WMMA B-operand bf16 layout -----------------
    pack_weights_kernel<<<dim3((NFEAT * NH) / 256, 1), 256, 0, stream>>>(
        w0, w0pk, NFEAT * NH);
    pack_weights_kernel<<<dim3((NH * NH) / 256, NL), 256, 0, stream>>>(
        convw, convpk, NH * NH);

    // --- input GEMMs ------------------------------------------------------
    input_gemm_kernel<<<N_NODES / 16, 256, 0, stream>>>(
        x, (const v16bf*)w0pk, b0, h0);
    input_gemm_kernel<<<N_NODES / 16, 256, 0, stream>>>(
        wild_feature, (const v16bf*)w0pk, b0, wh0);

    int zero_blocks = (N_NODES * NH) / 256;        // 25000
    int spmm_blocks = (N_EDGES * 32) / 256;        // 100000

    for (int br = 0; br < 2; ++br) {
        const float* base = br ? wh0 : h0;
        const int*   idx  = br ? wild_adj_idx : adj_idx;
        const float* val  = br ? wild_adj_val : adj_val;

        const float* cur = base;
        for (int l = 0; l < NL; ++l) {
            float theta = logf(LAMDA / (float)(l + 1) + 1.0f);
            float* nxt = (l & 1) ? pong : ping;
            const v16bf* Wpk =
                (const v16bf*)(convpk + (size_t)l * NH * NH);
            zero_kernel<<<zero_blocks, 256, 0, stream>>>(hi);
            spmm_kernel<<<spmm_blocks, 256, 0, stream>>>(idx, val, cur, hi);
            layer_update_kernel<<<N_NODES / 16, 256, 0, stream>>>(
                hi, base, cur, Wpk, theta, nxt);
            cur = nxt;
        }
        gather_mean_kernel<<<1, 128, 0, stream>>>(cur, site, out + 1 + br * NH);
    }

    head_kernel<<<1, 256, 0, stream>>>(aux, fc_w, fc_b, fc2_w, fc2_b,
                                       fc3_w, fc3_b, out);
}
